// GraphContrastiveEncoder_38165079392646
// MI455X (gfx1250) — compile-verified, matrix operand order checked by reading, not run on hardware
//
#include <hip/hip_runtime.h>
#include <hip/hip_bf16.h>
#include <stdint.h>

#define N_NODES 100000
#define N_EDGES 1600000
#define N_GRAPHS 64
#define HID 256
#define PROJ 128
#define NLAYERS 4

typedef __attribute__((ext_vector_type(16))) __bf16 v16bf;
typedef __attribute__((ext_vector_type(8)))  float  v8f;

__device__ __forceinline__ unsigned short f32_to_bf16_bits(float f) {
  unsigned int u = __builtin_bit_cast(unsigned int, f);
  u += 0x7FFFu + ((u >> 16) & 1u);            // round-to-nearest-even
  return (unsigned short)(u >> 16);
}

__device__ __forceinline__ void atomAddF(float* p, float v) {
  __hip_atomic_fetch_add(p, v, __ATOMIC_RELAXED, __HIP_MEMORY_SCOPE_AGENT);
}

// ---------------- generic utility kernels ----------------

__global__ void k_zero_f32(float* __restrict__ p, int n) {
  int i = blockIdx.x * blockDim.x + threadIdx.x;
  if (i < n) p[i] = 0.0f;
}

// Pack a [L,256,256] f32 weight into WMMA B-operand order, bf16:
// out[((l*16 + nt)*8 + ks)*32*16 + lane*16 + j] = W[l][k][col]
//   col = nt*16 + (lane&15) ;  k = ks*32 + ((j<8)?j:j+8) + (lane>>4)*8
__global__ void k_pack_B(const float* __restrict__ W,
                         unsigned short* __restrict__ out, int total) {
  int idx = blockIdx.x * blockDim.x + threadIdx.x;
  if (idx >= total) return;
  int l    = idx >> 16;          // 65536 per layer
  int t    = idx & 65535;
  int j    = t & 15;
  int lane = (t >> 4) & 31;
  int ks   = (t >> 9) & 7;
  int nt   = (t >> 12) & 15;
  int col  = nt * 16 + (lane & 15);
  int k    = ks * 32 + ((j < 8) ? j : j + 8) + (lane >> 4) * 8;
  out[idx] = f32_to_bf16_bits(W[(size_t)l * HID * HID + k * HID + col]);
}

// ---------------- graph aggregation ----------------

__global__ void k_degree(const int* __restrict__ ei, float* __restrict__ degf) {
  int e = blockIdx.x * blockDim.x + threadIdx.x;
  if (e < N_EDGES) atomAddF(&degf[ei[N_EDGES + e]], 1.0f);
}

__global__ void k_invdeg(float* __restrict__ degf) {
  int i = blockIdx.x * blockDim.x + threadIdx.x;
  if (i < N_NODES) degf[i] = 1.0f / fmaxf(degf[i], 1.0f);
}

// one block (256 threads) per edge: thread t handles feature t
__global__ __launch_bounds__(HID)
void k_scatter(const float* __restrict__ h, const int* __restrict__ ei,
               float* __restrict__ aggr) {
  int e = blockIdx.x;
  int t = threadIdx.x;
  int src = ei[e];
  int dst = ei[N_EDGES + e];
  atomAddF(&aggr[dst * HID + t], h[src * HID + t]);
}

// aggr *= 1/deg ; emit bf16 copies of aggr and h for the WMMA GEMM
__global__ void k_finish_aggr(const float* __restrict__ aggr,
                              const float* __restrict__ invdeg,
                              const float* __restrict__ h,
                              unsigned short* __restrict__ aggrB,
                              unsigned short* __restrict__ hB, int total) {
  int i = blockIdx.x * blockDim.x + threadIdx.x;
  if (i < total) {
    float a = aggr[i] * invdeg[i >> 8];    // i / HID
    aggrB[i] = f32_to_bf16_bits(a);
    hB[i]    = f32_to_bf16_bits(h[i]);
  }
}

// ---------------- WMMA GEMM: out = A0@B0 + A1@B1 + bias  (K=256 each) ----------
// One wave computes a 16x64 output strip: 4 accumulators reuse each A fragment.
// B operands come pre-packed in lane order -> one contiguous 32B load per frag.

__global__ __launch_bounds__(256)
void k_gemm_wmma(const unsigned short* __restrict__ A0,    // [N,256] bf16 (aggr)
                 const unsigned short* __restrict__ A1,    // [N,256] bf16 (h)
                 const unsigned short* __restrict__ B0p,   // packed [16][8][32][16]
                 const unsigned short* __restrict__ B1p,   // packed
                 const float* __restrict__ bias,           // [256]
                 float* __restrict__ out) {                // [N,256] f32
  int wave = blockIdx.x * (blockDim.x >> 5) + (threadIdx.x >> 5);
  int lane = threadIdx.x & 31;
  int mt = wave >> 2;            // 6250 m-tiles
  int ng = wave & 3;             // n-group: n-tiles ng*4 .. ng*4+3
  if (mt * 16 >= N_NODES) return;
  int half = lane >> 4;
  int r    = lane & 15;
  const int arow = mt * 16 + r;

  v8f c0 = {}, c1 = {}, c2 = {}, c3 = {};

  for (int ks = 0; ks < 8; ++ks) {
    v16bf a;
#pragma unroll
    for (int j = 0; j < 16; ++j) {
      int k = ks * 32 + ((j < 8) ? j : j + 8) + half * 8;
      a[j] = __builtin_bit_cast(__bf16, A0[arow * HID + k]);
    }
    const unsigned short* bp =
        B0p + ((((ng * 4) * 8 + ks) * 32 + lane) << 4);
    v16bf b0 = *(const v16bf*)(bp);
    v16bf b1 = *(const v16bf*)(bp + (8 * 32 * 16));
    v16bf b2 = *(const v16bf*)(bp + 2 * (8 * 32 * 16));
    v16bf b3 = *(const v16bf*)(bp + 3 * (8 * 32 * 16));
    c0 = __builtin_amdgcn_wmma_f32_16x16x32_bf16(false, a, false, b0, (short)0, c0, false, false);
    c1 = __builtin_amdgcn_wmma_f32_16x16x32_bf16(false, a, false, b1, (short)0, c1, false, false);
    c2 = __builtin_amdgcn_wmma_f32_16x16x32_bf16(false, a, false, b2, (short)0, c2, false, false);
    c3 = __builtin_amdgcn_wmma_f32_16x16x32_bf16(false, a, false, b3, (short)0, c3, false, false);
  }
  for (int ks = 0; ks < 8; ++ks) {
    v16bf a;
#pragma unroll
    for (int j = 0; j < 16; ++j) {
      int k = ks * 32 + ((j < 8) ? j : j + 8) + half * 8;
      a[j] = __builtin_bit_cast(__bf16, A1[arow * HID + k]);
    }
    const unsigned short* bp =
        B1p + ((((ng * 4) * 8 + ks) * 32 + lane) << 4);
    v16bf b0 = *(const v16bf*)(bp);
    v16bf b1 = *(const v16bf*)(bp + (8 * 32 * 16));
    v16bf b2 = *(const v16bf*)(bp + 2 * (8 * 32 * 16));
    v16bf b3 = *(const v16bf*)(bp + 3 * (8 * 32 * 16));
    c0 = __builtin_amdgcn_wmma_f32_16x16x32_bf16(false, a, false, b0, (short)0, c0, false, false);
    c1 = __builtin_amdgcn_wmma_f32_16x16x32_bf16(false, a, false, b1, (short)0, c1, false, false);
    c2 = __builtin_amdgcn_wmma_f32_16x16x32_bf16(false, a, false, b2, (short)0, c2, false, false);
    c3 = __builtin_amdgcn_wmma_f32_16x16x32_bf16(false, a, false, b3, (short)0, c3, false, false);
  }

  // D layout: VGPR rr -> M = rr + 8*half ; N = lane&15
  v8f cc[4] = {c0, c1, c2, c3};
#pragma unroll
  for (int q = 0; q < 4; ++q) {
    int n = (ng * 4 + q) * 16 + r;
    float bn = bias[n];
#pragma unroll
    for (int rr = 0; rr < 8; ++rr) {
      int m = mt * 16 + half * 8 + rr;
      out[m * HID + n] = cc[q][rr] + bn;
    }
  }
}

// ---------------- BatchNorm ----------------

__global__ __launch_bounds__(HID)
void k_bn_stats(const float* __restrict__ hp, float* __restrict__ colsum,
                float* __restrict__ colsumsq) {
  int col = threadIdx.x;
  int r0 = blockIdx.x * 256;
  int r1 = r0 + 256; if (r1 > N_NODES) r1 = N_NODES;
  float s = 0.0f, ss = 0.0f;
  for (int r = r0; r < r1; ++r) {
    float v = hp[r * HID + col];
    s += v; ss += v * v;
  }
  atomAddF(&colsum[col], s);
  atomAddF(&colsumsq[col], ss);
}

__global__ void k_bn_apply(const float* __restrict__ hp,
                           const float* __restrict__ colsum,
                           const float* __restrict__ colsumsq,
                           const float* __restrict__ gamma,
                           const float* __restrict__ beta,
                           float* __restrict__ h, int total) {
  int i = blockIdx.x * blockDim.x + threadIdx.x;
  if (i < total) {
    int col = i & (HID - 1);
    const float invN = 1.0f / (float)N_NODES;
    float mu  = colsum[col] * invN;
    float var = colsumsq[col] * invN - mu * mu;
    float y = (hp[i] - mu) * rsqrtf(var + 1e-5f) * gamma[col] + beta[col];
    h[i] = fmaxf(y, 0.0f);
  }
}

// ---------------- pooling + projection head ----------------

__global__ __launch_bounds__(HID)
void k_pool(const float* __restrict__ h, const int* __restrict__ batch,
            float* __restrict__ gsum, float* __restrict__ gcnt) {
  int node = blockIdx.x;
  int t = threadIdx.x;
  int g = batch[node];
  atomAddF(&gsum[g * HID + t], h[node * HID + t]);
  if (t == 0) atomAddF(&gcnt[g], 1.0f);
}

__global__ __launch_bounds__(HID)
void k_head(const float* __restrict__ gsum, const float* __restrict__ gcnt,
            const float* __restrict__ W1, const float* __restrict__ b1,
            const float* __restrict__ W2, const float* __restrict__ b2,
            float* __restrict__ out) {
  __shared__ float gm[HID];
  __shared__ float hid[HID];
  __shared__ float zs[PROJ];
  __shared__ float red[64];
  __shared__ float nrm;
  int g = blockIdx.x, t = threadIdx.x;
  float ic = 1.0f / fmaxf(gcnt[g], 1.0f);
  gm[t] = gsum[g * HID + t] * ic;
  __syncthreads();
  float acc = b1[t];
  for (int k = 0; k < HID; ++k) acc += gm[k] * W1[k * HID + t];
  hid[t] = fmaxf(acc, 0.0f);
  __syncthreads();
  if (t < PROJ) {
    float z = b2[t];
    for (int k = 0; k < HID; ++k) z += hid[k] * W2[k * PROJ + t];
    zs[t] = z;
  }
  __syncthreads();
  if (t < 64) red[t] = zs[t] * zs[t] + zs[t + 64] * zs[t + 64];
  __syncthreads();
  if (t < 32) red[t] += red[t + 32];
  __syncthreads();
  if (t < 16) red[t] += red[t + 16];
  __syncthreads();
  if (t < 8) red[t] += red[t + 8];
  __syncthreads();
  if (t < 4) red[t] += red[t + 4];
  __syncthreads();
  if (t < 2) red[t] += red[t + 2];
  __syncthreads();
  if (t == 0) nrm = fmaxf(sqrtf(red[0] + red[1]), 1e-12f);
  __syncthreads();
  if (t < PROJ) out[g * PROJ + t] = zs[t] / nrm;
}

// ---------------- host ----------------

static inline size_t alignup(size_t x) { return (x + 255) & ~(size_t)255; }

extern "C" void kernel_launch(void* const* d_in, const int* in_sizes, int n_in,
                              void* d_out, int out_size, void* d_ws, size_t ws_size,
                              hipStream_t stream) {
  const float* x     = (const float*)d_in[0];
  const int*   ei    = (const int*)d_in[1];
  const int*   batch = (const int*)d_in[2];
  const float* Wl    = (const float*)d_in[3];   // [L,256,256]
  const float* bl    = (const float*)d_in[4];   // [L,256]
  const float* Wr    = (const float*)d_in[5];   // [L,256,256]
  const float* gamma = (const float*)d_in[6];   // [L,256]
  const float* beta  = (const float*)d_in[7];   // [L,256]
  const float* W1    = (const float*)d_in[8];
  const float* b1    = (const float*)d_in[9];
  const float* W2    = (const float*)d_in[10];
  const float* b2    = (const float*)d_in[11];
  float* out = (float*)d_out;

  const size_t NF   = (size_t)N_NODES * HID;          // 25.6M elems
  const size_t NFb  = NF * sizeof(float);
  const size_t NF2b = NF * sizeof(unsigned short);
  const size_t WLb  = (size_t)NLAYERS * HID * HID * sizeof(unsigned short);

  char* ws = (char*)d_ws;
  size_t off = 0;
  float* h             = (float*)(ws + off); off += alignup(NFb);
  float* aggr          = (float*)(ws + off); off += alignup(NFb);   // reused as h_pre
  unsigned short* aggrB= (unsigned short*)(ws + off); off += alignup(NF2b);
  unsigned short* hB   = (unsigned short*)(ws + off); off += alignup(NF2b);
  float* invdeg        = (float*)(ws + off); off += alignup((size_t)N_NODES * 4);
  unsigned short* wlB  = (unsigned short*)(ws + off); off += alignup(WLb);
  unsigned short* wrB  = (unsigned short*)(ws + off); off += alignup(WLb);
  float* colsum        = (float*)(ws + off); off += alignup(HID * 4);
  float* colsumsq      = (float*)(ws + off); off += alignup(HID * 4);
  float* gsum          = (float*)(ws + off); off += alignup((size_t)N_GRAPHS * HID * 4);
  float* gcnt          = (float*)(ws + off); off += alignup((size_t)N_GRAPHS * 4);
  (void)ws_size; (void)in_sizes; (void)n_in; (void)out_size;

  const int T = 256;
  const int nfBlocks = (int)((NF + T - 1) / T);

  // h = x
  hipMemcpyAsync(h, x, NFb, hipMemcpyDeviceToDevice, stream);

  // weights -> bf16, pre-swizzled into WMMA B-operand lane order (once)
  {
    int nw = NLAYERS * HID * HID;
    k_pack_B<<<(nw + T - 1) / T, T, 0, stream>>>(Wl, wlB, nw);
    k_pack_B<<<(nw + T - 1) / T, T, 0, stream>>>(Wr, wrB, nw);
  }

  // 1/deg (once; edge structure is layer-invariant)
  k_zero_f32<<<(N_NODES + T - 1) / T, T, 0, stream>>>(invdeg, N_NODES);
  k_degree<<<(N_EDGES + T - 1) / T, T, 0, stream>>>(ei, invdeg);
  k_invdeg<<<(N_NODES + T - 1) / T, T, 0, stream>>>(invdeg);

  // 6250 m-tiles * 4 n-groups = 25000 waves ; 8 waves/block
  const int gemmBlocks = (N_NODES / 16) * 4 / 8;
  const int bnBlocks   = (N_NODES + 255) / 256;
  const size_t WPL = (size_t)HID * HID;   // packed weight elems per layer

  for (int layer = 0; layer < NLAYERS; ++layer) {
    k_zero_f32<<<nfBlocks, T, 0, stream>>>(aggr, (int)NF);
    k_scatter<<<N_EDGES, HID, 0, stream>>>(h, ei, aggr);
    k_finish_aggr<<<nfBlocks, T, 0, stream>>>(aggr, invdeg, h, aggrB, hB, (int)NF);
    // aggr (f32) no longer needed -> reuse as GEMM output h_pre
    k_gemm_wmma<<<gemmBlocks, 256, 0, stream>>>(
        aggrB, hB,
        wlB + (size_t)layer * WPL,
        wrB + (size_t)layer * WPL,
        bl + (size_t)layer * HID,
        aggr);
    k_zero_f32<<<1, T, 0, stream>>>(colsum, HID);
    k_zero_f32<<<1, T, 0, stream>>>(colsumsq, HID);
    k_bn_stats<<<bnBlocks, HID, 0, stream>>>(aggr, colsum, colsumsq);
    k_bn_apply<<<nfBlocks, T, 0, stream>>>(aggr, colsum, colsumsq,
                                           gamma + (size_t)layer * HID,
                                           beta + (size_t)layer * HID,
                                           h, (int)NF);
  }

  // graph pooling
  k_zero_f32<<<(N_GRAPHS * HID + T - 1) / T, T, 0, stream>>>(gsum, N_GRAPHS * HID);
  k_zero_f32<<<1, T, 0, stream>>>(gcnt, N_GRAPHS);
  k_pool<<<N_NODES, HID, 0, stream>>>(h, batch, gsum, gcnt);

  // projection head + L2 normalize
  k_head<<<N_GRAPHS, HID, 0, stream>>>(gsum, gcnt, W1, b1, W2, b2, out);
}
